// MultiHeadAttention_36369783063373
// MI455X (gfx1250) — matrix operand order repaired and checked by reading, not compiled, not run on hardware
//
#include <hip/hip_runtime.h>

typedef __bf16 bf16;
typedef __bf16 v16bf __attribute__((ext_vector_type(16)));
typedef __bf16 v8bf  __attribute__((ext_vector_type(8)));
typedef float  v8f   __attribute__((ext_vector_type(8)));
typedef int    v4i   __attribute__((ext_vector_type(4)));

typedef __attribute__((address_space(1))) v4i gv4i;  // global 128-bit chunk
typedef __attribute__((address_space(3))) v4i lv4i;  // LDS 128-bit chunk

#if __has_builtin(__builtin_amdgcn_global_load_async_to_lds_b128) && \
    __has_builtin(__builtin_amdgcn_s_wait_asynccnt)
#define USE_ASYNC 1
#else
#define USE_ASYNC 0
#endif

#if USE_ASYNC
__device__ __forceinline__ void async_copy16(const bf16* g, bf16* l) {
  __builtin_amdgcn_global_load_async_to_lds_b128((gv4i*)g, (lv4i*)l, 0, 0);
}
#endif

#define WMMA_BF16(a, b, c) \
  __builtin_amdgcn_wmma_f32_16x16x32_bf16(false, (a), false, (b), (short)0, (c), false, false)

static constexpr int Bb = 2, S = 2048, D = 1024, H = 16, HD = 64;
static constexpr int MROWS = Bb * S;  // 4096

// ---------------- conversion kernels ----------------

__global__ void k_cvt_x(const float* __restrict__ x, bf16* __restrict__ xh, int n) {
  int i = blockIdx.x * blockDim.x + threadIdx.x;
  if (i < n) xh[i] = (bf16)x[i];
}

// wT[n][k] = w[k][n], f32 -> bf16 (4 weights via blockIdx.z)
__global__ void k_cvt_wT(const float* __restrict__ w0, const float* __restrict__ w1,
                         const float* __restrict__ w2, const float* __restrict__ w3,
                         bf16* __restrict__ o0, bf16* __restrict__ o1,
                         bf16* __restrict__ o2, bf16* __restrict__ o3) {
  const float* w;
  bf16* o;
  switch (blockIdx.z) {
    case 0: w = w0; o = o0; break;
    case 1: w = w1; o = o1; break;
    case 2: w = w2; o = o2; break;
    default: w = w3; o = o3; break;
  }
  int idx = blockIdx.x * blockDim.x + threadIdx.x;  // n*D + k
  int k = idx & (D - 1);
  int n = idx >> 10;
  o[idx] = (bf16)w[k * D + n];
}

// Load a 16x32 bf16 A fragment from a row-major [*, D] matrix.
__device__ __forceinline__ v16bf load_a_frag(const bf16* rowptr, int kk, int hiSel) {
  const v8bf lo = *(const v8bf*)(rowptr + kk + hiSel * 16);      // K {0-7 | 16-23}
  const v8bf hi = *(const v8bf*)(rowptr + kk + 8 + hiSel * 16);  // K {8-15 | 24-31}
  v16bf a;
#pragma unroll
  for (int i = 0; i < 8; ++i) { a[i] = lo[i]; a[8 + i] = hi[i]; }
  return a;
}

// ---------------- fused QKV projection ----------------
// 256 threads = 8 waves; each wave computes a 32x64 tile (2 A frags, 8 accs).
// The whole block shares one 64-col B panel, cooperatively staged into LDS
// with async copies (double-buffered) when available.
// sel 0->Q, 1->K (row-major head split [B,H,S,64]), 2->V (transposed [B,H,64,S]).
__global__ void k_proj(const bf16* __restrict__ xh,
                       const bf16* __restrict__ wqT, const bf16* __restrict__ wkT,
                       const bf16* __restrict__ wvT,
                       const float* __restrict__ bq, const float* __restrict__ bk,
                       const float* __restrict__ bv,
                       bf16* __restrict__ Qh, bf16* __restrict__ Kh, bf16* __restrict__ VhT) {
  const int sel = blockIdx.z;
  const bf16* wT = (sel == 0) ? wqT : (sel == 1) ? wkT : wvT;
  const float* bias = (sel == 0) ? bq : (sel == 1) ? bk : bv;

  const int tid = threadIdx.x;
  const int lane = tid & 31;
  const int wave = tid >> 5;
  const int hiSel = lane >> 4;
  const int mb = blockIdx.x * 256 + wave * 32;
  const int nb = blockIdx.y * 64;

  v8f acc[2][4];
#pragma unroll
  for (int rr = 0; rr < 2; ++rr)
#pragma unroll
    for (int j = 0; j < 4; ++j)
#pragma unroll
      for (int r = 0; r < 8; ++r) acc[rr][j][r] = 0.0f;

  const bf16* arow0 = xh + (size_t)(mb + (lane & 15)) * D;
  const bf16* arow1 = arow0 + (size_t)16 * D;

#if USE_ASYNC
  // B panel: 64 cols x 32 k, layout [col][k]; each thread owns one 16B slice.
  __shared__ __attribute__((aligned(64))) bf16 bpanel[2][64 * 32];
  const int scol = tid >> 2, schunk = tid & 3;
  const bf16* sptr = wT + (size_t)(nb + scol) * D + schunk * 8;
  bf16* dp0 = &bpanel[0][scol * 32 + schunk * 8];
  bf16* dp1 = &bpanel[1][scol * 32 + schunk * 8];
  async_copy16(sptr, dp0);
  int cur = 0;
#endif

  for (int kk = 0; kk < D; kk += 32) {
#if USE_ASYNC
    if (kk + 32 < D) {
      async_copy16(sptr + kk + 32, cur ? dp0 : dp1);
      __builtin_amdgcn_s_wait_asynccnt(1);
    } else {
      __builtin_amdgcn_s_wait_asynccnt(0);
    }
    __syncthreads();
#endif
    const v16bf a0 = load_a_frag(arow0, kk, hiSel);
    const v16bf a1 = load_a_frag(arow1, kk, hiSel);
#pragma unroll
    for (int j = 0; j < 4; ++j) {
#if USE_ASYNC
      const v16bf bfrag =
          *(const v16bf*)(&bpanel[cur][(j * 16 + (lane & 15)) * 32 + hiSel * 16]);
#else
      const v16bf bfrag =
          *(const v16bf*)(wT + (size_t)(nb + j * 16 + (lane & 15)) * D + kk + hiSel * 16);
#endif
      acc[0][j] = WMMA_BF16(a0, bfrag, acc[0][j]);
      acc[1][j] = WMMA_BF16(a1, bfrag, acc[1][j]);
    }
#if USE_ASYNC
    __syncthreads();  // all waves done reading before next overwrite
    cur ^= 1;
#endif
  }

#pragma unroll
  for (int rr = 0; rr < 2; ++rr)
#pragma unroll
    for (int j = 0; j < 4; ++j) {
      const int ncol = nb + j * 16 + (lane & 15);
      const float bsv = bias[ncol];
      const int h = ncol >> 6, d = ncol & 63;
#pragma unroll
      for (int r = 0; r < 8; ++r) {
        const int mrow = mb + rr * 16 + (hiSel ? 8 + r : r);
        const int bidx = mrow >> 11, sidx = mrow & (S - 1);
        const bf16 hv = (bf16)(acc[rr][j][r] + bsv);
        if (sel == 0)
          Qh[(((size_t)bidx * H + h) * S + sidx) * HD + d] = hv;
        else if (sel == 1)
          Kh[(((size_t)bidx * H + h) * S + sidx) * HD + d] = hv;
        else
          VhT[(((size_t)bidx * H + h) * HD + d) * S + sidx] = hv;
      }
    }
}

// ---------------- flash attention ----------------
// grid: (S/128, B*H), 256 threads = 8 independent waves, each owns 16 q rows.
__global__ void k_attn(const bf16* __restrict__ Qh, const bf16* __restrict__ Kh,
                       const bf16* __restrict__ VhT, bf16* __restrict__ ah) {
  __shared__ __attribute__((aligned(32))) bf16 lds[8 * 16 * 32];
  const int lane = threadIdx.x & 31;
  const int wave = threadIdx.x >> 5;
  const int hiSel = lane >> 4;
  const int bh = blockIdx.y;
  const int b = bh >> 4, h = bh & 15;
  const int qbase = blockIdx.x * 128 + wave * 16;

  const bf16* Qp = Qh + (size_t)bh * S * HD;
  const bf16* Kp = Kh + (size_t)bh * S * HD;
  const bf16* Vp = VhT + (size_t)bh * HD * S;
  bf16* ldsW = lds + wave * (16 * 32);

  // resident Q A-fragments (16 rows x 64 d) -> two 16x32 fragments
  v16bf aq[2];
  {
    const bf16* qr = Qp + (size_t)(qbase + (lane & 15)) * HD;
#pragma unroll
    for (int ks = 0; ks < 2; ++ks) aq[ks] = load_a_frag(qr, ks * 32, hiSel);
  }

  v8f acc[4];
  float m_r[8], l_r[8];
#pragma unroll
  for (int j = 0; j < 4; ++j)
#pragma unroll
    for (int r = 0; r < 8; ++r) acc[j][r] = 0.0f;
#pragma unroll
  for (int r = 0; r < 8; ++r) { m_r[r] = -1e30f; l_r[r] = 0.0f; }

  const float scale = 0.125f;  // 1/sqrt(64)
  const int kmax = qbase + 16; // causal: only blocks with kb <= max q row

  for (int kb = 0; kb < kmax; kb += 32) {
    // prefetch next key block's K rows and V rows into cache
    if (kb + 32 < kmax) {
      __builtin_prefetch(Kp + (size_t)(kb + 32 + lane) * HD, 0, 3);
      __builtin_prefetch(Vp + (size_t)lane * S + kb + 32, 0, 3);
      __builtin_prefetch(Vp + (size_t)(lane + 32) * S + kb + 32, 0, 3);
    }

    const bool needMask = (kb + 31 > qbase);

    // scores: two 16x16 C tiles, K-dim = 64 (2 bf16 wmma steps each)
    v8f sc[2];
#pragma unroll
    for (int t = 0; t < 2; ++t) {
#pragma unroll
      for (int r = 0; r < 8; ++r) sc[t][r] = 0.0f;
      const int key = kb + t * 16 + (lane & 15);
      const bf16* kr = Kp + (size_t)key * HD;
#pragma unroll
      for (int ks = 0; ks < 2; ++ks) {
        const v16bf kfrag = *(const v16bf*)(kr + ks * 32 + hiSel * 16);
        sc[t] = WMMA_BF16(aq[ks], kfrag, sc[t]);
      }
    }

    // online softmax update (rows live in 16-lane halves)
    float pr[2][8];
#pragma unroll
    for (int r = 0; r < 8; ++r) {
      const int qrow = qbase + (hiSel ? 8 + r : r);
      float s0 = sc[0][r] * scale;
      float s1 = sc[1][r] * scale;
      if (needMask) {
        const int key0 = kb + (lane & 15);
        if (key0 > qrow) s0 = -1e30f;
        if (key0 + 16 > qrow) s1 = -1e30f;
      }
      float tm = fmaxf(s0, s1);
      tm = fmaxf(tm, __shfl_xor(tm, 1, 32));
      tm = fmaxf(tm, __shfl_xor(tm, 2, 32));
      tm = fmaxf(tm, __shfl_xor(tm, 4, 32));
      tm = fmaxf(tm, __shfl_xor(tm, 8, 32));
      const float mnew = fmaxf(m_r[r], tm);
      const float corr = __expf(m_r[r] - mnew);
      const float p0 = __expf(s0 - mnew);
      const float p1 = __expf(s1 - mnew);
      float rs = p0 + p1;
      rs += __shfl_xor(rs, 1, 32);
      rs += __shfl_xor(rs, 2, 32);
      rs += __shfl_xor(rs, 4, 32);
      rs += __shfl_xor(rs, 8, 32);
      l_r[r] = l_r[r] * corr + rs;
      m_r[r] = mnew;
#pragma unroll
      for (int j = 0; j < 4; ++j) acc[j][r] *= corr;
      pr[0][r] = p0;
      pr[1][r] = p1;
    }

    // C layout -> A layout via per-wave LDS strip (16x32 bf16)
#pragma unroll
    for (int t = 0; t < 2; ++t)
#pragma unroll
      for (int r = 0; r < 8; ++r) {
        const int rrow = hiSel ? 8 + r : r;
        ldsW[rrow * 32 + t * 16 + (lane & 15)] = (bf16)pr[t][r];
      }
    v16bf ap;
    {
      const bf16* pp = ldsW + (lane & 15) * 32 + hiSel * 16;
      const v8bf lo = *(const v8bf*)(pp);
      const v8bf hi = *(const v8bf*)(pp + 8);
#pragma unroll
      for (int i = 0; i < 8; ++i) { ap[i] = lo[i]; ap[8 + i] = hi[i]; }
    }

    // acc += P @ V  (B fragments contiguous from transposed V)
#pragma unroll
    for (int j = 0; j < 4; ++j) {
      const int d = j * 16 + (lane & 15);
      const v16bf vfrag = *(const v16bf*)(Vp + (size_t)d * S + kb + hiSel * 16);
      acc[j] = WMMA_BF16(ap, vfrag, acc[j]);
    }
  }

  // epilogue: normalize and store merged [B,S,D] bf16
#pragma unroll
  for (int r = 0; r < 8; ++r) {
    const float inv = 1.0f / l_r[r];
    const int qrow = qbase + (hiSel ? 8 + r : r);
    const size_t rowoff = ((size_t)b * S + qrow) * D + h * HD;
#pragma unroll
    for (int j = 0; j < 4; ++j) {
      const int d = j * 16 + (lane & 15);
      ah[rowoff + d] = (bf16)(acc[j][r] * inv);
    }
  }
}

// ---------------- output projection ----------------
// Same 32x64-per-wave blocking + async-staged B panel; f32 output + bias.
__global__ void k_out(const bf16* __restrict__ ah, const bf16* __restrict__ woT,
                      const float* __restrict__ bo, float* __restrict__ out) {
  const int tid = threadIdx.x;
  const int lane = tid & 31;
  const int wave = tid >> 5;
  const int hiSel = lane >> 4;
  const int mb = blockIdx.x * 256 + wave * 32;
  const int nb = blockIdx.y * 64;

  v8f acc[2][4];
#pragma unroll
  for (int rr = 0; rr < 2; ++rr)
#pragma unroll
    for (int j = 0; j < 4; ++j)
#pragma unroll
      for (int r = 0; r < 8; ++r) acc[rr][j][r] = 0.0f;

  const bf16* arow0 = ah + (size_t)(mb + (lane & 15)) * D;
  const bf16* arow1 = arow0 + (size_t)16 * D;

#if USE_ASYNC
  __shared__ __attribute__((aligned(64))) bf16 bpanel[2][64 * 32];
  const int scol = tid >> 2, schunk = tid & 3;
  const bf16* sptr = woT + (size_t)(nb + scol) * D + schunk * 8;
  bf16* dp0 = &bpanel[0][scol * 32 + schunk * 8];
  bf16* dp1 = &bpanel[1][scol * 32 + schunk * 8];
  async_copy16(sptr, dp0);
  int cur = 0;
#endif

  for (int kk = 0; kk < D; kk += 32) {
#if USE_ASYNC
    if (kk + 32 < D) {
      async_copy16(sptr + kk + 32, cur ? dp0 : dp1);
      __builtin_amdgcn_s_wait_asynccnt(1);
    } else {
      __builtin_amdgcn_s_wait_asynccnt(0);
    }
    __syncthreads();
#endif
    const v16bf a0 = load_a_frag(arow0, kk, hiSel);
    const v16bf a1 = load_a_frag(arow1, kk, hiSel);
#pragma unroll
    for (int j = 0; j < 4; ++j) {
#if USE_ASYNC
      const v16bf bfrag =
          *(const v16bf*)(&bpanel[cur][(j * 16 + (lane & 15)) * 32 + hiSel * 16]);
#else
      const v16bf bfrag =
          *(const v16bf*)(woT + (size_t)(nb + j * 16 + (lane & 15)) * D + kk + hiSel * 16);
#endif
      acc[0][j] = WMMA_BF16(a0, bfrag, acc[0][j]);
      acc[1][j] = WMMA_BF16(a1, bfrag, acc[1][j]);
    }
#if USE_ASYNC
    __syncthreads();
    cur ^= 1;
#endif
  }

#pragma unroll
  for (int rr = 0; rr < 2; ++rr)
#pragma unroll
    for (int j = 0; j < 4; ++j) {
      const int ncol = nb + j * 16 + (lane & 15);
      const float bsv = bo[ncol];
#pragma unroll
      for (int r = 0; r < 8; ++r) {
        const int mrow = mb + rr * 16 + (hiSel ? 8 + r : r);
        out[(size_t)mrow * D + ncol] = acc[rr][j][r] + bsv;
      }
    }
}

// ---------------- launcher ----------------
extern "C" void kernel_launch(void* const* d_in, const int* in_sizes, int n_in,
                              void* d_out, int out_size, void* d_ws, size_t ws_size,
                              hipStream_t stream) {
  (void)in_sizes; (void)n_in; (void)out_size; (void)ws_size;
  const float* x  = (const float*)d_in[0];
  // d_in[1] = mask: causal structure applied analytically, never read
  const float* wq = (const float*)d_in[2];
  const float* bq = (const float*)d_in[3];
  const float* wk = (const float*)d_in[4];
  const float* bk = (const float*)d_in[5];
  const float* wv = (const float*)d_in[6];
  const float* bv = (const float*)d_in[7];
  const float* wo = (const float*)d_in[8];
  const float* bo = (const float*)d_in[9];
  float* out = (float*)d_out;

  char* ws = (char*)d_ws;
  const size_t MB = 1024 * 1024;
  bf16* xh  = (bf16*)(ws + 0);        //  8 MB  [4096,1024]
  bf16* wqT = (bf16*)(ws + 8 * MB);   //  2 MB  [N,K]
  bf16* wkT = (bf16*)(ws + 10 * MB);
  bf16* wvT = (bf16*)(ws + 12 * MB);
  bf16* woT = (bf16*)(ws + 14 * MB);
  bf16* Qh  = (bf16*)(ws + 16 * MB);  //  8 MB  [B,H,S,64]
  bf16* Kh  = (bf16*)(ws + 24 * MB);  //  8 MB  [B,H,S,64]
  bf16* VhT = (bf16*)(ws + 32 * MB);  //  8 MB  [B,H,64,S]
  bf16* ah  = (bf16*)(ws + 40 * MB);  //  8 MB  [4096,1024]

  k_cvt_x<<<(Bb * S * D + 255) / 256, 256, 0, stream>>>(x, xh, Bb * S * D);
  dim3 gW(D * D / 256, 1, 4);
  k_cvt_wT<<<gW, 256, 0, stream>>>(wq, wk, wv, wo, wqT, wkT, wvT, woT);
  dim3 gP(MROWS / 256, D / 64, 3);
  k_proj<<<gP, 256, 0, stream>>>(xh, wqT, wkT, wvT, bq, bk, bv, Qh, Kh, VhT);
  dim3 gA(S / 128, Bb * H);
  k_attn<<<gA, 256, 0, stream>>>(Qh, Kh, VhT, ah);
  dim3 gO(MROWS / 256, D / 64);
  k_out<<<gO, 256, 0, stream>>>(ah, woT, bo, out);
}
